// MultiHeadSpikingAttention_37572373905507
// MI455X (gfx1250) — matrix operand order repaired and checked by reading, not compiled, not run on hardware
//
#include <hip/hip_runtime.h>
#include <hip/hip_bf16.h>

// ---------------------------------------------------------------------------
// MultiHeadSpikingAttention for MI455X (gfx1250), wave32 + WMMA bf16.
// out  = (x@wq+bq, x@wk+bk, x@wv+bv) -> relu(QK^T/8) -> @V -> @wo+bo
// d_out = [ out (B*S*D f32) | attn (B*H*S*S f32) ]
// Memory-bound on the 512MB attn write; GEMMs via v_wmma_f32_16x16x32_bf16,
// tile staging via global_load_async_to_lds_b128 + ASYNCcnt double-buffering.
// Workspace requirement: 48 MiB.
// ---------------------------------------------------------------------------

typedef __attribute__((ext_vector_type(16))) __bf16 v16bf;
typedef __attribute__((ext_vector_type(8)))  float  v8f;

#define B_     2
#define S_     2048
#define D_     1024
#define H_     16
#define DEPTH_ 64
#define BS_    (B_ * S_)          // 4096 rows

// ---- helpers ---------------------------------------------------------------

__device__ __forceinline__ unsigned short f2bf(float f) {
  union { float f; unsigned int u; } a; a.f = f;
  unsigned int u = a.u;
  u += 0x7FFFu + ((u >> 16) & 1u);      // round-to-nearest-even
  return (unsigned short)(u >> 16);
}

union FragU { v16bf v; uint4 q[2]; };

// A-fragment (16-bit, 16x32): lanes 0-15 row M=lane hold K0-15;
// lanes 16-31 same rows hold K16-31 (contiguous, matches load_frag_bt).
__device__ __forceinline__ v16bf load_frag_a(const unsigned short* p, int stride) {
  const int lane = threadIdx.x & 31;
  const int row  = lane & 15;
  const int hi   = lane >> 4;
  const unsigned short* s = p + row * stride + hi * 16;
  FragU u;
  u.q[0] = *(const uint4*)(s);        // K = hi*16 .. hi*16+7
  u.q[1] = *(const uint4*)(s + 8);    // K = hi*16+8 .. hi*16+15
  return u.v;
}

// B-fragment from a transposed tile Bt[n][k] (row = N, contiguous K):
// lanes 0-15 hold K0-15 of column N=lane, lanes 16-31 hold K16-31.
__device__ __forceinline__ v16bf load_frag_bt(const unsigned short* p, int stride) {
  const int lane = threadIdx.x & 31;
  const int row  = lane & 15;
  const int hi   = lane >> 4;
  const unsigned short* s = p + row * stride + hi * 16;
  FragU u;
  u.q[0] = *(const uint4*)(s);
  u.q[1] = *(const uint4*)(s + 8);
  return u.v;
}

__device__ __forceinline__ v8f wmma_bf16(v16bf a, v16bf b, v8f c) {
  return __builtin_amdgcn_wmma_f32_16x16x32_bf16(
      /*neg_a=*/false, a, /*neg_b=*/false, b,
      /*c_mod=*/(short)0, c, /*reuse_a=*/false, /*reuse_b=*/false);
}

// Async global -> LDS copy (16B per lane), tracked by ASYNCcnt.
// lds_off: workgroup-relative LDS byte address (= low 32 bits of generic ptr).
__device__ __forceinline__ void async_ld_b128(unsigned lds_off, const void* g) {
  asm volatile("global_load_async_to_lds_b128 %0, %1, off"
               :: "v"(lds_off), "v"(g) : "memory");
}
__device__ __forceinline__ void wait_async0() {
  asm volatile("s_wait_asynccnt 0x0" ::: "memory");
}
__device__ __forceinline__ unsigned lds_addr(const void* p) {
  return (unsigned)(unsigned long long)p;   // low 32 bits = LDS byte offset
}

// ---- phase 0: f32 -> bf16 conversion of x ----------------------------------

__global__ __launch_bounds__(256) void cvt_bf16_kernel(
    const float* __restrict__ in, unsigned short* __restrict__ out, int n4) {
  int i = (blockIdx.x * 256 + threadIdx.x);
  if (i < n4) {
    float4 f = *(const float4*)(in + (size_t)i * 4);
    union { unsigned short s[4]; uint2 u; } o;
    o.s[0] = f2bf(f.x); o.s[1] = f2bf(f.y);
    o.s[2] = f2bf(f.z); o.s[3] = f2bf(f.w);
    *(uint2*)(out + (size_t)i * 4) = o.u;
  }
}

// ---- phase 0b: weight transpose + convert: w[K,N] f32 -> wt[N,K] bf16 ------

__global__ __launch_bounds__(256) void cvt_wt_kernel(
    const float* __restrict__ w, unsigned short* __restrict__ wt) {
  __shared__ unsigned short tile[32][33];
  const int tx = blockIdx.x & 31;        // tile col (n / 32)
  const int ty = blockIdx.x >> 5;        // tile row (k / 32)
  const int lx = threadIdx.x & 31;
  const int ly = threadIdx.x >> 5;       // 0..7
#pragma unroll
  for (int j = 0; j < 4; ++j) {
    int kk = ly + j * 8;
    tile[kk][lx] = f2bf(w[(size_t)(ty * 32 + kk) * D_ + tx * 32 + lx]);
  }
  __syncthreads();
#pragma unroll
  for (int j = 0; j < 4; ++j) {
    int nn = ly + j * 8;
    wt[(size_t)(tx * 32 + nn) * D_ + ty * 32 + lx] = tile[lx][nn];
  }
}

// ---- WMMA GEMM: C[M,N] = A[M,K](bf16) * Wt[N,K](bf16)^T + bias -------------
// headSplit=1: write bf16 to [B,H,S,DEPTH]; headSplit=0: write f32 [M,N].
// Double-buffered LDS, async direct-to-LDS staging.

#define GSTR 40   // LDS row stride in halves (80B: 16B-aligned, bank-skewed)

__global__ __launch_bounds__(256) void gemm_wmma_kernel(
    const unsigned short* __restrict__ A,    // [M,K] bf16
    const unsigned short* __restrict__ Wt,   // [N,K] bf16
    const float* __restrict__ bias,          // [N] f32
    unsigned short* __restrict__ outH,       // head-split bf16
    float* __restrict__ outF,                // [M,N] f32
    int K, int headSplit) {
  __shared__ unsigned short sA[2][128 * GSTR];
  __shared__ unsigned short sB[2][128 * GSTR];

  const int tid  = threadIdx.x;
  const int wave = tid >> 5;
  const int lane = tid & 31;
  const int mbase = blockIdx.y * 128;
  const int nbase = blockIdx.x * 128;
  const int mr = (wave >> 1) * 32;     // wave sub-tile: 32 rows
  const int nc = (wave & 1) * 64;      //               64 cols
  const int rowInLane = lane >> 4;
  const int colInLane = lane & 15;

  v8f acc[2][4] = {};

  const int lr = tid >> 1;             // 0..127 staging row
  const int lc = (tid & 1) * 16;       // 0 / 16 staging col
  const unsigned short* gA = A  + (size_t)(mbase + lr) * K + lc;
  const unsigned short* gB = Wt + (size_t)(nbase + lr) * K + lc;

  auto stage = [&](int buf, int kt) {
    unsigned la = lds_addr(&sA[buf][lr * GSTR + lc]);
    async_ld_b128(la,      gA + kt);
    async_ld_b128(la + 16, gA + kt + 8);
    unsigned lb = lds_addr(&sB[buf][lr * GSTR + lc]);
    async_ld_b128(lb,      gB + kt);
    async_ld_b128(lb + 16, gB + kt + 8);
  };

  stage(0, 0);
  const int nIter = K / 32;
  for (int it = 0; it < nIter; ++it) {
    const int cur = it & 1;
    wait_async0();            // sA/sB[cur] fully landed in LDS
    __syncthreads();          // visible to all waves; prev buf free
    if (it + 1 < nIter) stage(cur ^ 1, (it + 1) * 32);

    v16bf af[2], bf[4];
#pragma unroll
    for (int i = 0; i < 2; ++i)
      af[i] = load_frag_a(&sA[cur][(mr + 16 * i) * GSTR], GSTR);
#pragma unroll
    for (int j = 0; j < 4; ++j)
      bf[j] = load_frag_bt(&sB[cur][(nc + 16 * j) * GSTR], GSTR);
#pragma unroll
    for (int i = 0; i < 2; ++i)
#pragma unroll
      for (int j = 0; j < 4; ++j)
        acc[i][j] = wmma_bf16(af[i], bf[j], acc[i][j]);
  }

  // epilogue: bias + store (C layout: elem v -> M = (lane/16)*8+v, N = lane%16)
#pragma unroll
  for (int i = 0; i < 2; ++i) {
#pragma unroll
    for (int j = 0; j < 4; ++j) {
      const int n = nbase + nc + 16 * j + colInLane;
      const float bv = bias[n];
#pragma unroll
      for (int v = 0; v < 8; ++v) {
        const int m = mbase + mr + 16 * i + rowInLane * 8 + v;
        const float val = acc[i][j][v] + bv;
        if (headSplit) {
          const int h = n >> 6, dd = n & 63;
          const int b = m >> 11, s = m & (S_ - 1);
          outH[(((size_t)(b * H_ + h)) * S_ + s) * DEPTH_ + dd] = f2bf(val);
        } else {
          outF[(size_t)m * D_ + n] = val;
        }
      }
    }
  }
}

// ---- fused attention: logits = QK^T/8, relu -> attn out; ctx = attn @ V ----
// K tiles async double-buffered; V tiles loaded to regs, scattered transposed.

#define ASTR 72   // LDS row stride in halves (144B)

__global__ __launch_bounds__(256) void attn_wmma_kernel(
    const unsigned short* __restrict__ Qh,   // [B*H, S, 64] bf16
    const unsigned short* __restrict__ Kh,
    const unsigned short* __restrict__ Vh,
    unsigned short* __restrict__ ctxH,       // [B, S, D] bf16
    float* __restrict__ attnOut) {           // [B*H, S, S] f32
  __shared__ unsigned short sQ[64 * ASTR];
  __shared__ unsigned short sK[2][64 * ASTR];
  __shared__ unsigned short sVt[2][64 * ASTR]; // transposed: [d][k-row]
  __shared__ unsigned short sP[64 * ASTR];     // relu'd logits, bf16

  const int tid  = threadIdx.x;
  const int wave = tid >> 5;
  const int lane = tid & 31;
  const int bh = blockIdx.y;
  const int b  = bh >> 4;
  const int h  = bh & 15;
  const int qbase = blockIdx.x * 64;

  const unsigned short* qptr = Qh + ((size_t)bh * S_ + qbase) * DEPTH_;
  const unsigned short* kpH  = Kh + (size_t)bh * S_ * DEPTH_;
  const unsigned short* vpH  = Vh + (size_t)bh * S_ * DEPTH_;
  float* attnHead = attnOut + (size_t)bh * S_ * S_;

  const int r = tid >> 2;           // staging row 0..63
  const int c = (tid & 3) * 16;     // staging col 0/16/32/48
  const unsigned short* kRow = kpH + (size_t)r * DEPTH_ + c;
  const unsigned short* vRow = vpH + (size_t)r * DEPTH_ + c;

  // wave -> two 16x16 tiles: rows tr..tr+15, cols tc..tc+31
  const int tr = (wave & 3) * 16;
  const int tc = (wave >> 2) * 32;
  const int rowInLane = lane >> 4;
  const int colInLane = lane & 15;

  // prologue: Q tile + K tile 0 async; V tile 0 via regs + transposed scatter
  {
    unsigned lq = lds_addr(&sQ[r * ASTR + c]);
    async_ld_b128(lq,      qptr + (size_t)r * DEPTH_ + c);
    async_ld_b128(lq + 16, qptr + (size_t)r * DEPTH_ + c + 8);
    unsigned lk = lds_addr(&sK[0][r * ASTR + c]);
    async_ld_b128(lk,      kRow);
    async_ld_b128(lk + 16, kRow + 8);
    union { uint4 q; unsigned short s[8]; } u0, u1;
    u0.q = *(const uint4*)(vRow);
    u1.q = *(const uint4*)(vRow + 8);
#pragma unroll
    for (int i = 0; i < 8; ++i) sVt[0][(c + i) * ASTR + r]     = u0.s[i];
#pragma unroll
    for (int i = 0; i < 8; ++i) sVt[0][(c + 8 + i) * ASTR + r] = u1.s[i];
  }

  v8f cacc[2] = {};
  const float scale = 0.125f;   // 1/sqrt(64)
  uint4 vreg0 = {}, vreg1 = {};

  for (int i = 0; i < S_ / 64; ++i) {
    const int cur  = i & 1;
    const int kb   = i * 64;
    const bool more = (i + 1) < (S_ / 64);
    wait_async0();          // sK[cur] (and sQ on i==0) landed
    __syncthreads();        // LDS visible; prev-iter readers done

    if (more) {             // stage next K async, next V into regs
      const unsigned short* kN = kRow + (size_t)(i + 1) * 64 * DEPTH_;
      unsigned lk = lds_addr(&sK[cur ^ 1][r * ASTR + c]);
      async_ld_b128(lk,      kN);
      async_ld_b128(lk + 16, kN + 8);
      const unsigned short* vN = vRow + (size_t)(i + 1) * 64 * DEPTH_;
      vreg0 = *(const uint4*)(vN);
      vreg1 = *(const uint4*)(vN + 8);
    }

    // logits tiles: A = Q rows, Bt = K rows (QK^T)
#pragma unroll
    for (int t = 0; t < 2; ++t) {
      const int nc2 = tc + t * 16;
      v8f lacc = {};
#pragma unroll
      for (int ks = 0; ks < 2; ++ks) {
        v16bf aq = load_frag_a (&sQ[tr * ASTR + ks * 32], ASTR);
        v16bf bk = load_frag_bt(&sK[cur][nc2 * ASTR + ks * 32], ASTR);
        lacc = wmma_bf16(aq, bk, lacc);
      }
#pragma unroll
      for (int v = 0; v < 8; ++v) {
        float x = lacc[v] * scale;
        x = x > 0.f ? x : 0.f;                 // ReLU attention
        const int qr = tr + rowInLane * 8 + v;
        const int kc = nc2 + colInLane;
        attnHead[(size_t)(qbase + qr) * S_ + (kb + kc)] = x;
        sP[qr * ASTR + kc] = f2bf(x);
      }
    }

    if (more) {             // transposed V scatter into next buffer
      union { uint4 q; unsigned short s[8]; } u0, u1;
      u0.q = vreg0; u1.q = vreg1;
#pragma unroll
      for (int j = 0; j < 8; ++j) sVt[cur ^ 1][(c + j) * ASTR + r]     = u0.s[j];
#pragma unroll
      for (int j = 0; j < 8; ++j) sVt[cur ^ 1][(c + 8 + j) * ASTR + r] = u1.s[j];
    }
    __syncthreads();        // sP complete

    // ctx accumulate: A = attn tile rows, Bt = sVt rows (cols = d)
#pragma unroll
    for (int t = 0; t < 2; ++t) {
      const int dc = tc + t * 16;
#pragma unroll
      for (int ks = 0; ks < 2; ++ks) {
        v16bf ap = load_frag_a (&sP[tr * ASTR + ks * 32], ASTR);
        v16bf bv = load_frag_bt(&sVt[cur][dc * ASTR + ks * 32], ASTR);
        cacc[t] = wmma_bf16(ap, bv, cacc[t]);
      }
    }
  }

  // write ctx (bf16) into concat layout [B,S,D]
#pragma unroll
  for (int t = 0; t < 2; ++t) {
    const int dc = tc + t * 16 + colInLane;
#pragma unroll
    for (int v = 0; v < 8; ++v) {
      const int qr = qbase + tr + rowInLane * 8 + v;
      ctxH[((size_t)b * S_ + qr) * D_ + h * DEPTH_ + dc] = f2bf(cacc[t][v]);
    }
  }
}

// ---------------------------------------------------------------------------

extern "C" void kernel_launch(void* const* d_in, const int* in_sizes, int n_in,
                              void* d_out, int out_size, void* d_ws, size_t ws_size,
                              hipStream_t stream) {
  (void)in_sizes; (void)n_in; (void)out_size; (void)ws_size;

  const float* x  = (const float*)d_in[0];
  const float* wq = (const float*)d_in[1];
  const float* bq = (const float*)d_in[2];
  const float* wk = (const float*)d_in[3];
  const float* bk = (const float*)d_in[4];
  const float* wv = (const float*)d_in[5];
  const float* bv = (const float*)d_in[6];
  const float* wo = (const float*)d_in[7];
  const float* bo = (const float*)d_in[8];

  float* outF = (float*)d_out;                       // [B,S,D]
  float* attn = outF + (size_t)BS_ * D_;             // [B,H,S,S]

  // workspace carve (bf16 halves), total 48 MiB
  unsigned short* xh   = (unsigned short*)d_ws;              // 4096x1024
  unsigned short* wqt  = xh  + (size_t)BS_ * D_;             // 1024x1024 (N,K)
  unsigned short* wkt  = wqt + (size_t)D_ * D_;
  unsigned short* wvt  = wkt + (size_t)D_ * D_;
  unsigned short* wot  = wvt + (size_t)D_ * D_;
  unsigned short* qh   = wot + (size_t)D_ * D_;              // [B*H,S,64]
  unsigned short* kh   = qh  + (size_t)BS_ * D_;
  unsigned short* vh   = kh  + (size_t)BS_ * D_;
  unsigned short* ctxh = vh  + (size_t)BS_ * D_;             // [B,S,D]

  // phase 0: conversions
  {
    int n4 = (BS_ * D_) / 4;
    cvt_bf16_kernel<<<dim3((n4 + 255) / 256), dim3(256), 0, stream>>>(x, xh, n4);
    dim3 tg(1024);  // (1024/32)^2 tiles
    cvt_wt_kernel<<<tg, dim3(256), 0, stream>>>(wq, wqt);
    cvt_wt_kernel<<<tg, dim3(256), 0, stream>>>(wk, wkt);
    cvt_wt_kernel<<<tg, dim3(256), 0, stream>>>(wv, wvt);
    cvt_wt_kernel<<<tg, dim3(256), 0, stream>>>(wo, wot);
  }

  // phase 1: Q/K/V projections (M=4096,N=1024,K=1024), head-split bf16 out
  {
    dim3 grid(D_ / 128, BS_ / 128);   // 8 x 32
    gemm_wmma_kernel<<<grid, dim3(256), 0, stream>>>(xh, wqt, bq, qh, nullptr, D_, 1);
    gemm_wmma_kernel<<<grid, dim3(256), 0, stream>>>(xh, wkt, bk, kh, nullptr, D_, 1);
    gemm_wmma_kernel<<<grid, dim3(256), 0, stream>>>(xh, wvt, bv, vh, nullptr, D_, 1);
  }

  // phase 2: fused relu-attention + context
  {
    dim3 grid(S_ / 64, B_ * H_);      // 32 x 32
    attn_wmma_kernel<<<grid, dim3(256), 0, stream>>>(qh, kh, vh, ctxh, attn);
  }

  // phase 3: output projection, f32 to d_out
  {
    dim3 grid(D_ / 128, BS_ / 128);
    gemm_wmma_kernel<<<grid, dim3(256), 0, stream>>>(ctxh, wot, bo, nullptr, outF, D_, 0);
  }
}
